// RingsNet_31207232373286
// MI455X (gfx1250) — matrix operand order repaired and checked by reading, not compiled, not run on hardware
//
#include <hip/hip_runtime.h>
#include <hip/hip_bf16.h>
#include <math.h>
#include <stdint.h>

// ---------------------------------------------------------------------------
// MI455X (gfx1250). GEMMs: v_wmma_f32_16x16x32_bf16, activations & weights in
// bf16 (weights converted+pre-transposed to [N][K] once per call). Tiles are
// staged global->LDS by the Tensor Data Mover (tensor_load_to_lds), double
// buffered with s_wait_tensorcnt, so the WMMA pipe overlaps the DMA.
// wave32; 256-thread blocks = 8 waves.
// ---------------------------------------------------------------------------

typedef __attribute__((ext_vector_type(16))) __bf16 v16bf;
typedef __attribute__((ext_vector_type(8)))  float  v8f;
typedef unsigned int v4u   __attribute__((ext_vector_type(4)));
typedef int          v8i_t __attribute__((ext_vector_type(8)));
typedef int          v4i_t __attribute__((ext_vector_type(4)));
typedef unsigned short u16;

#define D_HID   768
#define SEQL    64
#define NHEADS  12
#define HDIM    64
#define FFDIM   3072
#define NLAYERS 12
#define BATCH   256
#define MTOT    (BATCH * SEQL)   // 16384 token rows

union FragBF {
    v16bf v;
    uint4 u[2];
};

__device__ __forceinline__ u16 f2bf(float f) {
    unsigned int u = __float_as_uint(f);
    u += 0x7FFFu + ((u >> 16) & 1u);          // RNE
    return (u16)(u >> 16);
}
__device__ __forceinline__ float bf2f(u16 u) {
    return __uint_as_float((unsigned int)u << 16);
}

// ---------------------------------------------------------------------------
// TDM: async-DMA one bf16 tile of 128 rows x 64 K-elems from global to LDS.
// Descriptor per CDNA5 ISA ch.8: 2D tile, data_size=2B, LDS padding of 16B
// every 128B so LDS rows land at 144B stride (= u16[72], 16B-aligned chunks).
// Issued by one wave only; tracked with TENSORcnt.
// ---------------------------------------------------------------------------
__device__ __forceinline__ void tdm_load_tile_128x64(unsigned lds_off,
                                                     const void* gtile,
                                                     unsigned k_stride_elems)
{
    const unsigned long long ga = (unsigned long long)(uintptr_t)gtile;
    v4u g0;
    g0[0] = 1u;                                      // count=1 (valid D#)
    g0[1] = lds_off;                                 // lds_addr (bytes)
    g0[2] = (unsigned)ga;                            // global_addr[31:0]
    g0[3] = (unsigned)((ga >> 32) & 0x01FFFFFFu) | 0x80000000u; // [56:32] | type=2
    v8i_t g1;
    // workgroup_mask=0 | data_size=1(2B)<<16 | pad_enable<<20 |
    // pad_interval=4 (32 dwords = 128B) <<22 | pad_amount=3 (4 dwords = 16B) <<25
    g1[0] = (int)((1u << 16) | (1u << 20) | (4u << 22) | (3u << 25));
    const unsigned td0 = k_stride_elems;             // tensor_dim0 (K extent)
    const unsigned td1 = 0x100000u;                  // tensor_dim1 (large)
    g1[1] = (int)((td0 & 0xFFFFu) << 16);            // bits[63:48]=td0 lo
    g1[2] = (int)((td0 >> 16) | ((td1 & 0xFFFFu) << 16));
    g1[3] = (int)((td1 >> 16) | (64u << 16));        // tile_dim0 = 64 (K)
    g1[4] = (int)128u;                               // tile_dim1 = 128 rows, tile_dim2=0
    g1[5] = (int)k_stride_elems;                     // tensor_dim0_stride lo
    g1[6] = 0;                                       // stride0 hi | stride1 lo
    g1[7] = 0;
    v4i_t z4 = {0, 0, 0, 0};
#if defined(__clang_major__) && (__clang_major__ >= 23)
    v8i_t z8 = {0, 0, 0, 0, 0, 0, 0, 0};
    __builtin_amdgcn_tensor_load_to_lds(g0, g1, z4, z4, z8, 0);
#else
    __builtin_amdgcn_tensor_load_to_lds(g0, g1, z4, z4, 0);
#endif
}

__device__ __forceinline__ unsigned lds_off_of(const void* p) {
    return (unsigned)(uintptr_t)p;   // low 32 bits of LDS/flat-shared address
}

// ---------------------------------------------------------------------------
// GEMM: C[M,N] = epi(A[M,K]bf16 @ Wt[N,K]bf16^T + bias (+ res bf16))
// Block tile 128x128, staged K-step 64 (two WMMA substeps of 32).
// 8 waves; wave tile 32(M) x 64(N) -> 2x4 accumulators of v8f.
// EPI: 0 none, 1 tanh, 2 exact GELU.  OUTF32: store f32 (final) else bf16.
// ---------------------------------------------------------------------------
template<int EPI, bool RES, bool OUTF32>
__global__ __launch_bounds__(256)
void gemm_tdm_kernel(const u16* __restrict__ A, const u16* __restrict__ Wt,
                     const float* __restrict__ bias, const u16* __restrict__ res,
                     void* __restrict__ Cout, int M, int N, int K)
{
    __shared__ u16 As[2][128][72];   // [buf][m][k] rows at 144B (DMA-padded)
    __shared__ u16 Bs[2][128][72];   // [buf][n][k]

    const int tid  = threadIdx.x;
    const int lane = tid & 31;
    const int w    = tid >> 5;
    const int wm   = (w & 3) * 32;
    const int wn   = (w >> 2) * 64;
    const int bm   = blockIdx.y * 128;
    const int bn   = blockIdx.x * 128;
    const int half = lane >> 4;
    const int l15  = lane & 15;
    const int kfA  = half * 8;    // A lane: chunks at half*8 and half*8+16
    const int kfB  = half * 8;    // B lane: chunks at half*8 and half*8+16

    v8f acc[2][4];
#pragma unroll
    for (int i = 0; i < 2; ++i)
#pragma unroll
        for (int j = 0; j < 4; ++j)
#pragma unroll
            for (int e = 0; e < 8; ++e) acc[i][j][e] = 0.0f;

    const int nK = K >> 6;
    if (tid < 32) {
        tdm_load_tile_128x64(lds_off_of(&As[0][0][0]), A + (size_t)bm * K, K);
        tdm_load_tile_128x64(lds_off_of(&Bs[0][0][0]), Wt + (size_t)bn * K, K);
    }
    for (int it = 0; it < nK; ++it) {
        if (tid < 32) {
            if (it + 1 < nK) {
                const int nb = (it + 1) & 1;
                tdm_load_tile_128x64(lds_off_of(&As[nb][0][0]),
                                     A + (size_t)bm * K + (it + 1) * 64, K);
                tdm_load_tile_128x64(lds_off_of(&Bs[nb][0][0]),
                                     Wt + (size_t)bn * K + (it + 1) * 64, K);
                __builtin_amdgcn_s_wait_tensorcnt(2);  // tile `it` landed
            } else {
                __builtin_amdgcn_s_wait_tensorcnt(0);
            }
        }
        __syncthreads();
        const u16 (*Ab)[72] = As[it & 1];
        const u16 (*Bb)[72] = Bs[it & 1];
#pragma unroll
        for (int kk = 0; kk < 64; kk += 32) {
            FragBF a[2], b[4];
#pragma unroll
            for (int sm = 0; sm < 2; ++sm) {
                const u16* p = &Ab[wm + sm * 16 + l15][kk + kfA];
                a[sm].u[0] = *(const uint4*)p;
                a[sm].u[1] = *(const uint4*)(p + 16);
            }
#pragma unroll
            for (int sn = 0; sn < 4; ++sn) {
                const u16* p = &Bb[wn + sn * 16 + l15][kk + kfB];
                b[sn].u[0] = *(const uint4*)p;
                b[sn].u[1] = *(const uint4*)(p + 16);
            }
#pragma unroll
            for (int sm = 0; sm < 2; ++sm)
#pragma unroll
                for (int sn = 0; sn < 4; ++sn)
                    acc[sm][sn] = __builtin_amdgcn_wmma_f32_16x16x32_bf16(
                        false, a[sm].v, false, b[sn].v, (short)0, acc[sm][sn],
                        false, false);
        }
        __syncthreads();
    }

    // epilogue: C/D layout -> lane col n = l15, row m = r + 8*half
#pragma unroll
    for (int sm = 0; sm < 2; ++sm) {
#pragma unroll
        for (int sn = 0; sn < 4; ++sn) {
            const int n = bn + wn + sn * 16 + l15;
            const float bv = bias ? bias[n] : 0.0f;
#pragma unroll
            for (int r = 0; r < 8; ++r) {
                const int m = bm + wm + sm * 16 + half * 8 + r;
                float v = acc[sm][sn][r] + bv;
                if (RES) v += bf2f(res[(size_t)m * N + n]);
                if (EPI == 1) v = tanhf(v);
                else if (EPI == 2) v = 0.5f * v * (1.0f + erff(v * 0.70710678118f));
                if (OUTF32) ((float*)Cout)[(size_t)m * N + n] = v;
                else        ((u16*)Cout)[(size_t)m * N + n] = f2bf(v);
            }
        }
    }
}

// ---------------------------------------------------------------------------
// Attention: one block per (batch, head); bf16 Q/K/V 64x64 slices in LDS,
// scores and ctx on WMMA, f32 softmax in LDS.
// ---------------------------------------------------------------------------
__global__ __launch_bounds__(256)
void attn_kernel(const u16* __restrict__ Q, const u16* __restrict__ Kx,
                 const u16* __restrict__ V, u16* __restrict__ O)
{
    __shared__ u16   Qs[64][72];   // [query][dim]
    __shared__ u16   Ks[64][72];   // [key][dim]
    __shared__ u16   Vt[64][72];   // [dim][key]
    __shared__ float Ss[64][64];   // scores f32
    __shared__ u16   Ps[64][72];   // probs bf16 [query][key]

    const int b  = blockIdx.x;
    const int hh = blockIdx.y;
    const int tid  = threadIdx.x;
    const int lane = tid & 31;
    const int w    = tid >> 5;
    const int half = lane >> 4;
    const int l15  = lane & 15;
    const int kfA  = half * 8;
    const int kfB  = half * 8;

    const size_t base = ((size_t)b * SEQL) * D_HID + (size_t)hh * HDIM;

    for (int i = tid; i < 64 * 8; i += 256) {       // 512 x 16B per matrix
        int r = i >> 3, c = (i & 7) * 8;
        *(uint4*)&Qs[r][c] = *(const uint4*)(Q + base + (size_t)r * D_HID + c);
        *(uint4*)&Ks[r][c] = *(const uint4*)(Kx + base + (size_t)r * D_HID + c);
        uint4 vv = *(const uint4*)(V + base + (size_t)r * D_HID + c);
        const u16* pv = (const u16*)&vv;
#pragma unroll
        for (int j = 0; j < 8; ++j) Vt[c + j][r] = pv[j];
    }
    __syncthreads();

    // scores: 16 WMMA tiles (4x4), 2 per wave
#pragma unroll
    for (int tt = 0; tt < 2; ++tt) {
        const int t = w * 2 + tt;
        const int tm = (t >> 2) * 16, tn = (t & 3) * 16;
        v8f acc;
#pragma unroll
        for (int e = 0; e < 8; ++e) acc[e] = 0.0f;
#pragma unroll
        for (int kk = 0; kk < 64; kk += 32) {
            FragBF a, bb;
            const u16* pa = &Qs[tm + l15][kk + kfA];
            a.u[0] = *(const uint4*)pa;  a.u[1] = *(const uint4*)(pa + 16);
            const u16* pb = &Ks[tn + l15][kk + kfB];
            bb.u[0] = *(const uint4*)pb; bb.u[1] = *(const uint4*)(pb + 16);
            acc = __builtin_amdgcn_wmma_f32_16x16x32_bf16(
                false, a.v, false, bb.v, (short)0, acc, false, false);
        }
#pragma unroll
        for (int r = 0; r < 8; ++r)
            Ss[tm + half * 8 + r][tn + l15] = acc[r] * 0.125f;  // 1/sqrt(64)
    }
    __syncthreads();

    if (tid < 64) {   // softmax over keys, one row per thread
        float mx = -1e30f;
        for (int j = 0; j < 64; ++j) mx = fmaxf(mx, Ss[tid][j]);
        float s = 0.0f;
        for (int j = 0; j < 64; ++j) s += expf(Ss[tid][j] - mx);
        const float inv = 1.0f / s;
        for (int j = 0; j < 64; ++j)
            Ps[tid][j] = f2bf(expf(Ss[tid][j] - mx) * inv);
    }
    __syncthreads();

    // ctx = P[64,keys] @ V -> [query, dim]; B operand from Vt[dim][key]
#pragma unroll
    for (int tt = 0; tt < 2; ++tt) {
        const int t = w * 2 + tt;
        const int tm = (t >> 2) * 16, tn = (t & 3) * 16;
        v8f acc;
#pragma unroll
        for (int e = 0; e < 8; ++e) acc[e] = 0.0f;
#pragma unroll
        for (int kk = 0; kk < 64; kk += 32) {
            FragBF a, bb;
            const u16* pa = &Ps[tm + l15][kk + kfA];
            a.u[0] = *(const uint4*)pa;  a.u[1] = *(const uint4*)(pa + 16);
            const u16* pb = &Vt[tn + l15][kk + kfB];
            bb.u[0] = *(const uint4*)pb; bb.u[1] = *(const uint4*)(pb + 16);
            acc = __builtin_amdgcn_wmma_f32_16x16x32_bf16(
                false, a.v, false, bb.v, (short)0, acc, false, false);
        }
#pragma unroll
        for (int r = 0; r < 8; ++r)
            O[base + (size_t)(tm + half * 8 + r) * D_HID + tn + l15] =
                f2bf(acc[r]);
    }
}

// ---------------------------------------------------------------------------
// LayerNorm over D=768 (bf16 in/out, f32 math); block per token, 3 elems/thr.
// Optional fused add: + pos_emb[row%64] + temb[row/64] (both f32).
// ---------------------------------------------------------------------------
__global__ __launch_bounds__(256)
void ln_kernel(const u16* __restrict__ X, const float* __restrict__ pos,
               const float* __restrict__ tembp, const float* __restrict__ g,
               const float* __restrict__ beta, u16* __restrict__ Y)
{
    const int row = blockIdx.x;
    const int tid = threadIdx.x;
    const size_t rb = (size_t)row * D_HID;
    float v[3];
#pragma unroll
    for (int i = 0; i < 3; ++i) {
        const int c = tid + i * 256;
        float x = bf2f(X[rb + c]);
        if (pos) x += pos[(row & (SEQL - 1)) * D_HID + c]
                    + tembp[(size_t)(row >> 6) * D_HID + c];
        v[i] = x;
    }
    __shared__ float red[256];
    red[tid] = v[0] + v[1] + v[2];
    __syncthreads();
    for (int st = 128; st; st >>= 1) {
        if (tid < st) red[tid] += red[tid + st];
        __syncthreads();
    }
    const float mean = red[0] * (1.0f / 768.0f);
    __syncthreads();
    const float d0 = v[0] - mean, d1 = v[1] - mean, d2 = v[2] - mean;
    red[tid] = d0 * d0 + d1 * d1 + d2 * d2;
    __syncthreads();
    for (int st = 128; st; st >>= 1) {
        if (tid < st) red[tid] += red[tid + st];
        __syncthreads();
    }
    const float inv = rsqrtf(red[0] * (1.0f / 768.0f) + 1e-12f);
#pragma unroll
    for (int i = 0; i < 3; ++i) {
        const int c = tid + i * 256;
        Y[rb + c] = f2bf((v[i] - mean) * inv * g[c] + beta[c]);
    }
}

// ---------------------------------------------------------------------------
// Weight prep: W[K,N] f32 -> Wt[N,K] bf16 (transpose + convert), 32x32 tiles.
// ---------------------------------------------------------------------------
__global__ __launch_bounds__(256)
void wprep_kernel(const float* __restrict__ W, u16* __restrict__ Wt,
                  int K, int N)
{
    __shared__ float tile[32][33];
    const int bk = blockIdx.y * 32, bnn = blockIdx.x * 32;
    const int tx = threadIdx.x & 31, ty = threadIdx.x >> 5;
#pragma unroll
    for (int r = ty; r < 32; r += 8)
        tile[r][tx] = W[(size_t)(bk + r) * N + bnn + tx];
    __syncthreads();
#pragma unroll
    for (int r = ty; r < 32; r += 8)
        Wt[(size_t)(bnn + r) * K + bk + tx] = f2bf(tile[tx][r]);
}

// elementwise f32 -> bf16 (x_t)
__global__ __launch_bounds__(256)
void cvt_bf16_kernel(const float* __restrict__ X, u16* __restrict__ Y)
{
    const size_t i = ((size_t)blockIdx.x * 256 + threadIdx.x) * 4;
    const float4 f = *(const float4*)(X + i);
    Y[i + 0] = f2bf(f.x); Y[i + 1] = f2bf(f.y);
    Y[i + 2] = f2bf(f.z); Y[i + 3] = f2bf(f.w);
}

// ---------------------------------------------------------------------------
// Time embedding: sinusoid(128) -> Linear -> SiLU -> Linear  (f32, tiny)
// ---------------------------------------------------------------------------
__global__ __launch_bounds__(256)
void temb_kernel(const float* __restrict__ t, const float* __restrict__ w1,
                 const float* __restrict__ b1, const float* __restrict__ w2,
                 const float* __restrict__ b2, float* __restrict__ out)
{
    const int bb = blockIdx.x;
    const int tid = threadIdx.x;
    const float tv = t[bb];
    __shared__ float e[128];
    __shared__ float y1[256];
    if (tid < 64) {
        const float fr = expf(-logf(10000.0f) * (float)tid / 64.0f);
        const float a = tv * fr;
        e[tid] = cosf(a);
        e[tid + 64] = sinf(a);
    }
    __syncthreads();
    {
        float s = b1[tid];
        for (int i = 0; i < 128; ++i) s += e[i] * w1[i * 256 + tid];
        y1[tid] = s / (1.0f + expf(-s));   // SiLU
    }
    __syncthreads();
    for (int d = tid; d < D_HID; d += 256) {
        float s = b2[d];
        for (int j = 0; j < 256; ++j) s += y1[j] * w2[j * D_HID + d];
        out[(size_t)bb * D_HID + d] = s;
    }
}

// ---------------------------------------------------------------------------
extern "C" void kernel_launch(void* const* d_in, const int* in_sizes, int n_in,
                              void* d_out, int out_size, void* d_ws, size_t ws_size,
                              hipStream_t stream) {
    (void)in_sizes; (void)n_in; (void)out_size; (void)ws_size;

    const float* x_t     = (const float*)d_in[0];
    const float* t       = (const float*)d_in[1];
    const float* up_w1   = (const float*)d_in[2];
    const float* up_b1   = (const float*)d_in[3];
    const float* up_w2   = (const float*)d_in[4];
    const float* up_b2   = (const float*)d_in[5];
    const float* te_w1   = (const float*)d_in[6];
    const float* te_b1   = (const float*)d_in[7];
    const float* te_w2   = (const float*)d_in[8];
    const float* te_b2   = (const float*)d_in[9];
    const float* pos_emb = (const float*)d_in[10];
    const float* ln_w    = (const float*)d_in[11];
    const float* ln_b    = (const float*)d_in[12];
    const float* Wq      = (const float*)d_in[13];
    const float* bq      = (const float*)d_in[14];
    const float* Wk      = (const float*)d_in[15];
    const float* bk      = (const float*)d_in[16];
    const float* Wv      = (const float*)d_in[17];
    const float* bv      = (const float*)d_in[18];
    const float* Wo      = (const float*)d_in[19];
    const float* bo      = (const float*)d_in[20];
    const float* ln1_w   = (const float*)d_in[21];
    const float* ln1_b   = (const float*)d_in[22];
    const float* Wf1     = (const float*)d_in[23];
    const float* bf1     = (const float*)d_in[24];
    const float* Wf2     = (const float*)d_in[25];
    const float* bf2     = (const float*)d_in[26];
    const float* down_w1 = (const float*)d_in[27 + 2];
    const float* down_b1 = (const float*)d_in[30];
    const float* down_w2 = (const float*)d_in[31];
    const float* down_b2 = (const float*)d_in[32];
    const float* ln2_w   = (const float*)d_in[27];
    const float* ln2_b   = (const float*)d_in[28];

    char* p = (char*)d_ws;
    auto carve = [&](size_t bytes) -> char* {
        char* r = p;
        p += (bytes + 255) & ~(size_t)255;
        return r;
    };
    const size_t MD = (size_t)MTOT * D_HID;
    u16* h    = (u16*)carve(MD * 2);
    u16* tmp  = (u16*)carve(MD * 2);
    u16* qb   = (u16*)carve(MD * 2);
    u16* kb   = (u16*)carve(MD * 2);
    u16* vb   = (u16*)carve(MD * 2);
    u16* mid  = (u16*)carve((size_t)MTOT * FFDIM * 2);
    u16* xbf  = (u16*)carve((size_t)MTOT * 512 * 2);
    float* te = (float*)carve((size_t)BATCH * D_HID * 4);
    u16* up1t = (u16*)carve((size_t)768 * 512 * 2);
    u16* up2t = (u16*)carve((size_t)768 * 768 * 2);
    u16* wqT  = (u16*)carve((size_t)NLAYERS * 768 * 768 * 2);
    u16* wkT  = (u16*)carve((size_t)NLAYERS * 768 * 768 * 2);
    u16* wvT  = (u16*)carve((size_t)NLAYERS * 768 * 768 * 2);
    u16* woT  = (u16*)carve((size_t)NLAYERS * 768 * 768 * 2);
    u16* wf1T = (u16*)carve((size_t)NLAYERS * 3072 * 768 * 2);
    u16* wf2T = (u16*)carve((size_t)NLAYERS * 768 * 3072 * 2);
    u16* dw1t = (u16*)carve((size_t)768 * 768 * 2);
    u16* dw2t = (u16*)carve((size_t)256 * 768 * 2);

    const dim3 blk(256);
    const dim3 g768(D_HID / 128, MTOT / 128);
    const dim3 gFF(FFDIM / 128, MTOT / 128);
    const dim3 gOut(256 / 128, MTOT / 128);
    const dim3 tT768x768(768 / 32, 768 / 32);

    // ---- one-time prep (recomputed every call; stateless) ----
    cvt_bf16_kernel<<<(MTOT * 512) / 1024, blk, 0, stream>>>(x_t, xbf);
    wprep_kernel<<<dim3(768 / 32, 512 / 32), blk, 0, stream>>>(up_w1, up1t, 512, 768);
    wprep_kernel<<<tT768x768, blk, 0, stream>>>(up_w2, up2t, 768, 768);
    wprep_kernel<<<tT768x768, blk, 0, stream>>>(down_w1, dw1t, 768, 768);
    wprep_kernel<<<dim3(256 / 32, 768 / 32), blk, 0, stream>>>(down_w2, dw2t, 768, 256);
    for (int l = 0; l < NLAYERS; ++l) {
        const size_t wo_ = (size_t)l * 768 * 768;
        const size_t wf_ = (size_t)l * 768 * 3072;
        wprep_kernel<<<tT768x768, blk, 0, stream>>>(Wq + wo_, wqT + wo_, 768, 768);
        wprep_kernel<<<tT768x768, blk, 0, stream>>>(Wk + wo_, wkT + wo_, 768, 768);
        wprep_kernel<<<tT768x768, blk, 0, stream>>>(Wv + wo_, wvT + wo_, 768, 768);
        wprep_kernel<<<tT768x768, blk, 0, stream>>>(Wo + wo_, woT + wo_, 768, 768);
        wprep_kernel<<<dim3(3072 / 32, 768 / 32), blk, 0, stream>>>(
            Wf1 + wf_, wf1T + wf_, 768, 3072);
        wprep_kernel<<<dim3(768 / 32, 3072 / 32), blk, 0, stream>>>(
            Wf2 + wf_, wf2T + wf_, 3072, 768);
    }
    temb_kernel<<<BATCH, blk, 0, stream>>>(t, te_w1, te_b1, te_w2, te_b2, te);

    // ---- input_up_proj: tanh(x @ up_w1 + b) @ up_w2 + b ----
    gemm_tdm_kernel<1, false, false><<<g768, blk, 0, stream>>>(
        xbf, up1t, up_b1, nullptr, qb, MTOT, D_HID, 512);
    gemm_tdm_kernel<0, false, false><<<g768, blk, 0, stream>>>(
        qb, up2t, up_b2, nullptr, tmp, MTOT, D_HID, D_HID);
    ln_kernel<<<MTOT, blk, 0, stream>>>(tmp, pos_emb, te, ln_w, ln_b, h);

    for (int l = 0; l < NLAYERS; ++l) {
        const size_t wofs = (size_t)l * 768 * 768;
        const size_t bofs = (size_t)l * D_HID;
        gemm_tdm_kernel<0, false, false><<<g768, blk, 0, stream>>>(
            h, wqT + wofs, bq + bofs, nullptr, qb, MTOT, D_HID, D_HID);
        gemm_tdm_kernel<0, false, false><<<g768, blk, 0, stream>>>(
            h, wkT + wofs, bk + bofs, nullptr, kb, MTOT, D_HID, D_HID);
        gemm_tdm_kernel<0, false, false><<<g768, blk, 0, stream>>>(
            h, wvT + wofs, bv + bofs, nullptr, vb, MTOT, D_HID, D_HID);
        attn_kernel<<<dim3(BATCH, NHEADS), blk, 0, stream>>>(qb, kb, vb, tmp);
        gemm_tdm_kernel<0, true, false><<<g768, blk, 0, stream>>>(
            tmp, woT + wofs, bo + bofs, h, mid, MTOT, D_HID, D_HID);
        ln_kernel<<<MTOT, blk, 0, stream>>>(
            mid, nullptr, nullptr, ln1_w + bofs, ln1_b + bofs, h);
        gemm_tdm_kernel<2, false, false><<<gFF, blk, 0, stream>>>(
            h, wf1T + (size_t)l * 768 * 3072, bf1 + (size_t)l * FFDIM,
            nullptr, mid, MTOT, FFDIM, D_HID);
        gemm_tdm_kernel<0, true, false><<<g768, blk, 0, stream>>>(
            mid, wf2T + (size_t)l * 768 * 3072, bf2 + bofs, h, tmp,
            MTOT, D_HID, FFDIM);
        ln_kernel<<<MTOT, blk, 0, stream>>>(
            tmp, nullptr, nullptr, ln2_w + bofs, ln2_b + bofs, h);
    }

    // ---- output_down_proj: tanh(h @ dw1 + b) @ dw2 + b -> f32 [MTOT,256] ----
    gemm_tdm_kernel<1, false, false><<<g768, blk, 0, stream>>>(
        h, dw1t, down_b1, nullptr, tmp, MTOT, D_HID, D_HID);
    gemm_tdm_kernel<0, false, true><<<gOut, blk, 0, stream>>>(
        tmp, dw2t, down_b2, nullptr, d_out, MTOT, 256, D_HID);
}
